// SampledAct_79860621902199
// MI455X (gfx1250) — matrix operand order repaired and checked by reading, not compiled, not run on hardware
//
#include <hip/hip_runtime.h>
#include <hip/hip_bf16.h>

// CDNA5 WMMA vector types
typedef __attribute__((ext_vector_type(16))) __bf16 v16bf;
typedef __attribute__((ext_vector_type(8)))  __bf16 v8bf;
typedef __attribute__((ext_vector_type(8)))  float  v8f;

namespace {
constexpr int kB = 32, kT = 512, kD = 1024, kN = 50000, kS = 4096;
constexpr int RT   = 64;       // tokens (rows) per workgroup
constexpr int CT   = 128;      // sampled columns per tile
constexpr int KB   = 64;       // K-slab depth (two wmma k-steps of 32)
constexpr int XS   = kD + 16;  // x LDS row stride in halfs (2080B, 16B aligned)
constexpr int WSTR = KB + 16;  // W LDS col stride in halfs (160B, 32B aligned)
constexpr int NT   = 256;      // 8 wave32 waves
constexpr float NEG_INF = -1e9f;
}

// round-to-nearest-even f32 -> bf16, packed pair
__device__ __forceinline__ unsigned short f2bf(float f) {
  unsigned u = __float_as_uint(f);
  u += 0x7fffu + ((u >> 16) & 1u);
  return (unsigned short)(u >> 16);
}
__device__ __forceinline__ unsigned pack2(float a, float b) {
  return (unsigned)f2bf(a) | ((unsigned)f2bf(b) << 16);
}

// log-uniform candidate sampler: log E[count] under unique sampling
__device__ __forceinline__ float log_expected_count(int id) {
  const float invLogN1 = 1.0f / logf((float)kN + 1.0f);
  float p = log1pf(1.0f / ((float)id + 1.0f)) * invLogN1;
  // log(1 - (1-p)^S) = log(-expm1(S * log1p(-p)))
  return logf(-expm1f((float)kS * log1pf(-p)));
}

// gather CT sampled rows' [kb*KB, kb*KB+KB) slice of W into LDS as bf16,
// column-contiguous: wl[col*WSTR + kk]
__device__ __forceinline__ void fill_w_slab(unsigned short* __restrict__ wl,
                                            const float* __restrict__ W,
                                            const int* __restrict__ idsS,
                                            int kb, int tid) {
#pragma unroll
  for (int it = 0; it < (CT * KB / 4) / NT; ++it) {  // 8 float4 per thread
    int i   = tid + it * NT;
    int col = i >> 4;            // KB/4 = 16 chunks per column
    int c4  = i & 15;
    int id  = idsS[col];
    const float4 v = *(const float4*)(W + (size_t)id * kD + kb * KB + c4 * 4);
    uint2 p;
    p.x = pack2(v.x, v.y);
    p.y = pack2(v.z, v.w);
    *(uint2*)(wl + col * WSTR + c4 * 4) = p;
  }
}

__global__ __launch_bounds__(NT)
void sampled_softmax_kernel(const float* __restrict__ x,
                            const float* __restrict__ W,
                            const float* __restrict__ bias,
                            const int*   __restrict__ labels,
                            const int*   __restrict__ sampled,
                            float* __restrict__ partials) {
  // ~176 KB LDS total: legal on CDNA5 (320 KB / WGP), 1 workgroup resident.
  __shared__ alignas(32) unsigned short xs[RT * XS];        // 130 KB bf16 x tile
  __shared__ alignas(32) unsigned short wt[2][CT * WSTR];   // 40 KB dbl-buf W slab
  __shared__ int   idsS[CT];
  __shared__ float colBias[CT];
  __shared__ int   labS[RT];
  __shared__ float tlS[RT];
  __shared__ float rowSum[RT];
  __shared__ float redS[NT / 32];

  const int tid  = threadIdx.x;
  const int lane = tid & 31;
  const int wv   = tid >> 5;          // 0..7
  const int half = lane >> 4;         // 0/1 (lane 0-15 vs 16-31)
  const int ln   = lane & 15;
  const int bIdx = blockIdx.x >> 3;   // kT/RT = 8 row tiles per batch
  const int t0   = (blockIdx.x & 7) * RT;

  if (tid < RT) rowSum[tid] = 0.0f;

  // ---- stage x tile to LDS as bf16 (loaded once, reused by all 32 col tiles)
  const float* xbase = x + (size_t)(bIdx * kT + t0) * kD;
#pragma unroll 4
  for (int it = 0; it < (RT * kD / 4) / NT; ++it) {   // 64 float4 per thread
    int i  = tid + it * NT;
    int r  = i >> 8;          // kD/4 = 256 chunks per row
    int c4 = i & 255;
    const float4 v = *(const float4*)(xbase + (size_t)r * kD + c4 * 4);
    uint2 p;
    p.x = pack2(v.x, v.y);
    p.y = pack2(v.z, v.w);
    *(uint2*)(xs + r * XS + c4 * 4) = p;
  }

  // ---- true-class logits (64 length-1024 dots: trivial VALU work)
  for (int rr = 0; rr < 8; ++rr) {
    const int r   = wv * 8 + rr;
    const int lab = labels[bIdx * kT + t0 + r];
    const float* xr = xbase + (size_t)r * kD;
    const float* wr = W + (size_t)lab * kD;
    float acc = 0.0f;
    for (int d = lane; d < kD; d += 32) acc += xr[d] * wr[d];
    acc += __shfl_xor(acc, 16);
    acc += __shfl_xor(acc, 8);
    acc += __shfl_xor(acc, 4);
    acc += __shfl_xor(acc, 2);
    acc += __shfl_xor(acc, 1);
    if (lane == 0) {
      labS[r] = lab;
      tlS[r]  = acc + bias[lab] - log_expected_count(lab);
    }
  }
  __syncthreads();

  // wave -> 32x32 output patch
  const int m0 = (wv & 1) * 32;
  const int n0 = (wv >> 1) * 32;

  float rowAcc[2][8];                 // lane-local partial sum(exp) per row
#pragma unroll
  for (int dm = 0; dm < 2; ++dm)
#pragma unroll
    for (int i = 0; i < 8; ++i) rowAcc[dm][i] = 0.0f;

  const v8f vzero = {};

  for (int ct = 0; ct < kS / CT; ++ct) {            // 32 column tiles
    if (tid < CT) {
      int id = sampled[bIdx * kS + ct * CT + tid];
      idsS[tid]    = id;
      colBias[tid] = bias[id] - log_expected_count(id);
    }
    __syncthreads();

    v8f acc[2][2];
#pragma unroll
    for (int dm = 0; dm < 2; ++dm)
#pragma unroll
      for (int dn = 0; dn < 2; ++dn) acc[dm][dn] = vzero;

    fill_w_slab(&wt[0][0], W, idsS, 0, tid);

    for (int kbi = 0; kbi < kD / KB; ++kbi) {       // 16 K-slabs
      __syncthreads();
      if (kbi + 1 < kD / KB)
        fill_w_slab(&wt[(kbi + 1) & 1][0], W, idsS, kbi + 1, tid);

      const unsigned short* wl = &wt[kbi & 1][0];
      const int kbase = kbi * KB;
#pragma unroll
      for (int ks = 0; ks < 2; ++ks) {              // two 32-deep wmma k-steps
        const int klo = half * 8;
        v16bf a[2], bfr[2];
#pragma unroll
        for (int dm = 0; dm < 2; ++dm) {
          // A 16x32 bf16, k order this code uses: lanes0-15 row=ln K[0:16),
          // lanes16-31 same row K[16:32)
          const unsigned short* ap =
              xs + (m0 + dm * 16 + ln) * XS + kbase + ks * 32;
          v8bf alo = *(const v8bf*)(ap + 2 * klo);
          v8bf ahi = *(const v8bf*)(ap + 8 + 2 * klo);
          a[dm] = __builtin_shufflevector(alo, ahi, 0, 1, 2, 3, 4, 5, 6, 7, 8,
                                          9, 10, 11, 12, 13, 14, 15);
        }
#pragma unroll
        for (int dn = 0; dn < 2; ++dn) {
          // B 32x16 bf16: lane holds col=ln, K[half*16 .. +16) contiguous
          const unsigned short* bp =
              wl + (n0 + dn * 16 + ln) * WSTR + ks * 32 + half * 16;
          bfr[dn] = *(const v16bf*)bp;
        }
#pragma unroll
        for (int dm = 0; dm < 2; ++dm)
#pragma unroll
          for (int dn = 0; dn < 2; ++dn)
            acc[dm][dn] = __builtin_amdgcn_wmma_f32_16x16x32_bf16(
                false, a[dm], false, bfr[dn], (short)0, acc[dm][dn], false,
                false);
      }
    }

    // ---- fused softmax epilogue (stays in registers; no logits to DRAM)
    // C layout: vgpr i, lanes0-15 -> row m0+dm*16+i, lanes16-31 -> +8; col=ln
#pragma unroll
    for (int dm = 0; dm < 2; ++dm) {
      int labv[8];
#pragma unroll
      for (int i = 0; i < 8; ++i)
        labv[i] = labS[m0 + dm * 16 + half * 8 + i];
#pragma unroll
      for (int dn = 0; dn < 2; ++dn) {
        const int col  = n0 + dn * 16 + ln;
        const float cb = colBias[col];
        const int cid  = idsS[col];
#pragma unroll
        for (int i = 0; i < 8; ++i) {
          float v = acc[dm][dn][i] + cb;
          v = (cid == labv[i]) ? NEG_INF : v;   // remove accidental hits
          rowAcc[dm][i] += __expf(v);
        }
      }
    }
    __syncthreads();  // epilogue reads of idsS/colBias done before next tile
  }

  // ---- fold the 16 column-lanes of each row, then combine the 4 col-waves
#pragma unroll
  for (int dm = 0; dm < 2; ++dm)
#pragma unroll
    for (int i = 0; i < 8; ++i) {
      float v = rowAcc[dm][i];
      v += __shfl_xor(v, 1);
      v += __shfl_xor(v, 2);
      v += __shfl_xor(v, 4);
      v += __shfl_xor(v, 8);
      if (ln == 0) atomicAdd(&rowSum[m0 + dm * 16 + half * 8 + i], v);
    }
  __syncthreads();

  // ---- per-token loss: logsumexp([true, sampled]) - true_logit
  float part = 0.0f;
  if (tid < RT) {
    const float tl = tlS[tid];
    part = logf(rowSum[tid] + __expf(tl)) - tl;
  }
  part += __shfl_xor(part, 16);
  part += __shfl_xor(part, 8);
  part += __shfl_xor(part, 4);
  part += __shfl_xor(part, 2);
  part += __shfl_xor(part, 1);
  if (lane == 0) redS[wv] = part;
  __syncthreads();
  if (tid == 0) {
    float s = 0.0f;
#pragma unroll
    for (int i = 0; i < NT / 32; ++i) s += redS[i];
    partials[blockIdx.x] = s;
  }
}

__global__ __launch_bounds__(256)
void reduce_kernel(const float* __restrict__ partials, float* __restrict__ out,
                   int n) {
  __shared__ float sh[8];
  float s = 0.0f;
  for (int i = threadIdx.x; i < n; i += 256) s += partials[i];
  s += __shfl_xor(s, 16);
  s += __shfl_xor(s, 8);
  s += __shfl_xor(s, 4);
  s += __shfl_xor(s, 2);
  s += __shfl_xor(s, 1);
  if ((threadIdx.x & 31) == 0) sh[threadIdx.x >> 5] = s;
  __syncthreads();
  if (threadIdx.x == 0) {
    float t = 0.0f;
#pragma unroll
    for (int i = 0; i < 8; ++i) t += sh[i];
    out[0] = t * (0.5f / (float)(kB * kT));  // 0.5 * mean over B of mean over T
  }
}

extern "C" void kernel_launch(void* const* d_in, const int* in_sizes, int n_in,
                              void* d_out, int out_size, void* d_ws,
                              size_t ws_size, hipStream_t stream) {
  const float* x       = (const float*)d_in[0];
  const float* W       = (const float*)d_in[1];
  const float* b       = (const float*)d_in[2];
  const int*   labels  = (const int*)d_in[3];
  const int*   sampled = (const int*)d_in[4];
  float* partials = (float*)d_ws;            // 256 floats of scratch

  const int nblocks = kB * (kT / RT);        // 32 * 8 = 256 workgroups
  sampled_softmax_kernel<<<nblocks, NT, 0, stream>>>(x, W, b, labels, sampled,
                                                     partials);
  reduce_kernel<<<1, 256, 0, stream>>>(partials, (float*)d_out, nblocks);
}
